// EquivariantReactivityModel_1683627180841
// MI455X (gfx1250) — matrix operand order repaired and checked by reading, not compiled, not run on hardware
//
#include <hip/hip_runtime.h>

typedef __attribute__((ext_vector_type(16))) _Float16 v16h;
typedef __attribute__((ext_vector_type(8)))  _Float16 v8h;
typedef __attribute__((ext_vector_type(8)))  float    v8f;

static constexpr int N_ATOMS = 16380;
static constexpr int NPAD    = 16384;   // padded rows for 16-row WMMA tiles
static constexpr int KNN     = 16;
static constexpr int HDIM    = 64;
static constexpr int PLANE   = NPAD * HDIM;     // 2^20
static constexpr int RRES    = 819;
static constexpr int APR     = 20;
static constexpr int LAYERS  = 6;
static constexpr int HEADS   = 8;
static constexpr int BINS    = 100;
static constexpr int NCHUNK  = 4;               // kNN candidate chunks
static constexpr int CHSZ    = N_ATOMS / NCHUNK; // 4095

// epilogue selectors
static constexpr int EPI_NONE   = 0;
static constexpr int EPI_ACC    = 1;
static constexpr int EPI_GELU16 = 2;  // f16 store of gelu(x), A-swizzled for CH=4 consumer
static constexpr int EPI_SIG    = 3;

// ---------------------------------------------------------------------------
// A-fragment swizzle: (row, k) -> index in fragment-ordered f16 buffer.
// Any aligned 8-channel group (row, 8g..8g+7) maps to 8 contiguous halves.
// ---------------------------------------------------------------------------
__device__ __forceinline__ int aswz(int row, int k, int CH) {
    int tile = row >> 4, m = row & 15;
    int kc = k >> 5, r5 = k & 31;
    int hi = (r5 >> 3) & 1;
    int i  = (r5 & 7) | ((r5 >> 4) << 3);
    return (((tile * CH + kc) * 32) + hi * 16 + m) * 16 + i;
}

__device__ __forceinline__ void mad4(float4& a, float p, const float4& x) {
    a.x += p * x.x; a.y += p * x.y; a.z += p * x.z; a.w += p * x.w;
}
__device__ __forceinline__ void madd4(float4& a, float p, const float4& x,
                                      const float4& w, float d) {
    a.x += p * (x.x + w.x * d);
    a.y += p * (x.y + w.y * d);
    a.z += p * (x.z + w.z * d);
    a.w += p * (x.w + w.w * d);
}

// ---------------------------------------------------------------------------
// weight swizzle: (K x M) f32 row-major -> B-fragment-ordered f16
// ---------------------------------------------------------------------------
__global__ void wswz_kernel(const float* __restrict__ src, _Float16* __restrict__ dst,
                            int K, int M, int nbatch) {
    int idx = blockIdx.x * blockDim.x + threadIdx.x;
    int per = K * M;
    if (idx >= per * nbatch) return;
    int b = idx / per;
    int r = idx % per;
    int i    = r & 15;
    int lane = (r >> 4) & 31;
    int chunk = r >> 9;
    int CH = K >> 5;
    int kc = chunk % CH;
    int ct = chunk / CH;
    int krow = kc * 32 + 8 * (lane >> 4) + (i & 7) + 16 * (i >> 3);
    int col  = ct * 16 + (lane & 15);
    dst[idx] = (_Float16)src[(size_t)b * per + krow * M + col];
}

// concat Wq|Wk|Wv -> (L,64,192) f32 (then wswz'd)
__global__ void qkvcat_kernel(const float* __restrict__ Wq, const float* __restrict__ Wk,
                              const float* __restrict__ Wv, float* __restrict__ Wqkv) {
    int idx = blockIdx.x * blockDim.x + threadIdx.x;
    if (idx >= LAYERS * 64 * 192) return;
    int l = idx / (64 * 192);
    int rem = idx % (64 * 192);
    int k = rem / 192;
    int c = rem % 192;
    float v;
    if (c < 64)       v = Wq[l * 4096 + k * 64 + c];
    else if (c < 128) v = Wk[l * 4096 + k * 64 + (c - 64)];
    else              v = Wv[l * 4096 + k * 64 + (c - 128)];
    Wqkv[idx] = v;
}

// ---------------------------------------------------------------------------
// embedding gather: thread per (atom, 8-channel group); contiguous f16 store
// ---------------------------------------------------------------------------
__global__ void embed_kernel(const int* __restrict__ atom_ids, const int* __restrict__ res_ids,
                             const int* __restrict__ elem_ids,
                             const float* __restrict__ ea, const float* __restrict__ er,
                             const float* __restrict__ ee, _Float16* __restrict__ feat16) {
    int idx = blockIdx.x * blockDim.x + threadIdx.x;
    if (idx >= NPAD * 8) return;
    int n = idx >> 3, g = idx & 7;
    float v[8];
    if (n < N_ATOMS) {
        const int c0 = g * 8;
#pragma unroll
        for (int i = 0; i < 8; ++i) {
            const int c = c0 + i;
            float t;
            if (c < 32)      t = ea[atom_ids[n] * 32 + c];
            else if (c < 48) t = er[res_ids[n] * 16 + (c - 32)];
            else             t = ee[elem_ids[n] * 16 + (c - 48)];
            v[i] = t;
        }
    } else {
#pragma unroll
        for (int i = 0; i < 8; ++i) v[i] = 0.0f;
    }
    const int base = aswz(n, g * 8, 2);
#pragma unroll
    for (int i = 0; i < 8; ++i) feat16[base + i] = (_Float16)v[i];
}

// ---------------------------------------------------------------------------
// kNN stage 1: per (dst, chunk) thread keeps top-16 over its candidate range
// ---------------------------------------------------------------------------
__global__ __launch_bounds__(256) void knn_partial_kernel(const float* __restrict__ coords,
                                                          float* __restrict__ pd2,
                                                          int* __restrict__ pidx) {
    __shared__ float sx[256], sy[256], sz[256];
    const int i = blockIdx.x * 256 + threadIdx.x;
    const int chunk = blockIdx.y;
    const int lo = chunk * CHSZ;
    const bool act = i < N_ATOMS;
    const int ii = act ? i : 0;
    const float cx = coords[ii * 3 + 0];
    const float cy = coords[ii * 3 + 1];
    const float cz = coords[ii * 3 + 2];

    float bd[KNN];
    int   bi[KNN];
#pragma unroll
    for (int t = 0; t < KNN; ++t) { bd[t] = 3.0e38f; bi[t] = 0; }
    float worst = 3.0e38f;

    for (int base = lo; base < lo + CHSZ; base += 256) {
        int j = base + threadIdx.x;
        if (j < lo + CHSZ) {
            sx[threadIdx.x] = coords[j * 3 + 0];
            sy[threadIdx.x] = coords[j * 3 + 1];
            sz[threadIdx.x] = coords[j * 3 + 2];
        }
        __syncthreads();
        const int cnt = min(256, lo + CHSZ - base);
        for (int u = 0; u < cnt; ++u) {
            const int j2 = base + u;
            const float dx = cx - sx[u];
            const float dy = cy - sy[u];
            const float dz = cz - sz[u];
            const float d2 = dx * dx + dy * dy + dz * dz;
            if (j2 != i && d2 < worst) {
                int slot = 0; float mx = bd[0];
#pragma unroll
                for (int t = 1; t < KNN; ++t) if (bd[t] > mx) { mx = bd[t]; slot = t; }
#pragma unroll
                for (int t = 0; t < KNN; ++t) if (t == slot) { bd[t] = d2; bi[t] = j2; }
                worst = bd[0];
#pragma unroll
                for (int t = 1; t < KNN; ++t) worst = fmaxf(worst, bd[t]);
            }
        }
        __syncthreads();
    }
    if (act) {
#pragma unroll
        for (int t = 0; t < KNN; ++t) {
            pd2 [(i * NCHUNK + chunk) * KNN + t] = bd[t];
            pidx[(i * NCHUNK + chunk) * KNN + t] = bi[t];
        }
    }
}

// kNN stage 2: merge NCHUNK*16 partial candidates -> final 16
__global__ void knn_merge_kernel(const float* __restrict__ pd2, const int* __restrict__ pidx,
                                 int* __restrict__ nbr) {
    const int i = blockIdx.x * blockDim.x + threadIdx.x;
    if (i >= N_ATOMS) return;
    float bd[KNN];
    int   bi[KNN];
#pragma unroll
    for (int t = 0; t < KNN; ++t) { bd[t] = 3.0e38f; bi[t] = 0; }
    float worst = 3.0e38f;
    for (int u = 0; u < NCHUNK * KNN; ++u) {
        const float d2 = pd2[i * NCHUNK * KNN + u];
        const int   j  = pidx[i * NCHUNK * KNN + u];
        if (d2 < worst) {
            int slot = 0; float mx = bd[0];
#pragma unroll
            for (int t = 1; t < KNN; ++t) if (bd[t] > mx) { mx = bd[t]; slot = t; }
#pragma unroll
            for (int t = 0; t < KNN; ++t) if (t == slot) { bd[t] = d2; bi[t] = j; }
            worst = bd[0];
#pragma unroll
            for (int t = 1; t < KNN; ++t) worst = fmaxf(worst, bd[t]);
        }
    }
#pragma unroll
    for (int t = 0; t < KNN; ++t) nbr[i * KNN + t] = bi[t];
}

// ---------------------------------------------------------------------------
// edge geometry
// ---------------------------------------------------------------------------
__global__ void edges_kernel(const float* __restrict__ coords, const int* __restrict__ nbr,
                             float* __restrict__ dirn, int* __restrict__ bins) {
    int e = blockIdx.x * blockDim.x + threadIdx.x;
    if (e >= N_ATOMS * KNN) return;
    int n = e >> 4;
    int j = nbr[e];
    float dx = coords[n * 3 + 0] - coords[j * 3 + 0];
    float dy = coords[n * 3 + 1] - coords[j * 3 + 1];
    float dz = coords[n * 3 + 2] - coords[j * 3 + 2];
    float d = sqrtf(dx * dx + dy * dy + dz * dz + 1e-12f);
    float inv = 1.0f / d;
    dirn[e * 3 + 0] = dx * inv;
    dirn[e * 3 + 1] = dy * inv;
    dirn[e * 3 + 2] = dz * inv;
    int b = (int)(d * 10.0f);
    bins[e] = min(max(b, 0), BINS - 1);
}

// ---------------------------------------------------------------------------
// WMMA GEMM, LDS-free, software-pipelined: B fragments resident in registers;
// A fragments for a batch of row tiles prefetched before the WMMA chain.
// ---------------------------------------------------------------------------
template <int KDIM, int MDIM, int EPI>
__global__ __launch_bounds__(2 * MDIM) void gemm_wmma_kernel(const _Float16* __restrict__ A,
                                                             const _Float16* __restrict__ Bsw,
                                                             void* __restrict__ Out) {
    constexpr int CH = KDIM / 32;
    constexpr int RT = (CH <= 2) ? 4 : 2;   // row tiles per prefetch batch
    const int wave = threadIdx.x >> 5;
    const int lane = threadIdx.x & 31;
    const int m  = lane & 15;
    const int hi = lane >> 4;
    const int tile0 = blockIdx.x * 4;

    v16h bfrag[CH];
#pragma unroll
    for (int kc = 0; kc < CH; ++kc)
        bfrag[kc] = *(const v16h*)(Bsw + ((size_t)(wave * CH + kc) * 32 + lane) * 16);

    const int ocol = wave * 16 + m;

#pragma unroll
    for (int bt = 0; bt < 4 / RT; ++bt) {
        v16h afrag[RT][CH];
#pragma unroll
        for (int rt = 0; rt < RT; ++rt)
#pragma unroll
            for (int kc = 0; kc < CH; ++kc)
                afrag[rt][kc] = *(const v16h*)(
                    A + ((size_t)((tile0 + bt * RT + rt) * CH + kc) * 32 + lane) * 16);

#pragma unroll
        for (int rt = 0; rt < RT; ++rt) {
            v8f c = {};
#pragma unroll
            for (int kc = 0; kc < CH; ++kc)
                c = __builtin_amdgcn_wmma_f32_16x16x32_f16(false, afrag[rt][kc], false,
                                                           bfrag[kc], (short)0, c, false, false);
            const int tile = tile0 + bt * RT + rt;
            const int orow = tile * 16 + 8 * hi;
            if (EPI == EPI_GELU16) {
                _Float16* O = (_Float16*)Out;
#pragma unroll
                for (int r = 0; r < 8; ++r) {
                    float x = c[r];
                    float t = tanhf(0.7978845608f * (x + 0.044715f * x * x * x));
                    O[aswz(orow + r, ocol, MDIM / 32)] = (_Float16)(0.5f * x * (1.0f + t));
                }
            } else if (EPI == EPI_SIG) {
                float* O = (float*)Out;
#pragma unroll
                for (int r = 0; r < 8; ++r)
                    O[(size_t)(orow + r) * MDIM + ocol] = 1.0f / (1.0f + __expf(-c[r]));
            } else if (EPI == EPI_ACC) {
                float* O = (float*)Out;
#pragma unroll
                for (int r = 0; r < 8; ++r) O[(size_t)(orow + r) * MDIM + ocol] += c[r];
            } else {
                float* O = (float*)Out;
#pragma unroll
                for (int r = 0; r < 8; ++r) O[(size_t)(orow + r) * MDIM + ocol] = c[r];
            }
        }
    }
}

// ---------------------------------------------------------------------------
// LayerNorm, thread-per-row (64 ch in 16 float4 regs), vectorized in and out
// ---------------------------------------------------------------------------
__global__ __launch_bounds__(256) void ln16_kernel(const float* __restrict__ s,
                                                   _Float16* __restrict__ h16) {
    const int row = blockIdx.x * 256 + threadIdx.x;
    const float4* S4 = (const float4*)(s + (size_t)row * 64);
    float x[64];
    float sum = 0.0f;
#pragma unroll
    for (int g = 0; g < 16; ++g) {
        float4 t = S4[g];
        x[g * 4 + 0] = t.x; x[g * 4 + 1] = t.y; x[g * 4 + 2] = t.z; x[g * 4 + 3] = t.w;
        sum += t.x + t.y + t.z + t.w;
    }
    const float mean = sum * (1.0f / 64.0f);
    float var = 0.0f;
#pragma unroll
    for (int c = 0; c < 64; ++c) { const float d = x[c] - mean; var += d * d; }
    const float r = rsqrtf(var * (1.0f / 64.0f) + 1e-5f);
#pragma unroll
    for (int g = 0; g < 8; ++g) {
        const int base = aswz(row, g * 8, 2);
#pragma unroll
        for (int i = 0; i < 8; ++i)
            h16[base + i] = (_Float16)((x[g * 8 + i] - mean) * r);
    }
}

// ---------------------------------------------------------------------------
// fused attention, float4-vectorized gathers; msg stored as contiguous 8-half
// group in the A-swizzled layout
// ---------------------------------------------------------------------------
__global__ __launch_bounds__(256) void attn_kernel(const float* __restrict__ qkv,
                                                   const int* __restrict__ nbr,
                                                   const int* __restrict__ bins,
                                                   const float* __restrict__ dirn,
                                                   const float* __restrict__ dist_bias_l,
                                                   const float* __restrict__ wdir_l,
                                                   const float* __restrict__ vvals,
                                                   float* __restrict__ vbuf,
                                                   _Float16* __restrict__ msg16) {
    const int n = blockIdx.x * 32 + (threadIdx.x >> 3);
    const int head = threadIdx.x & 7;
    const int hd8 = head * 8;
    const int h4 = hd8 >> 2;
    const bool act = n < N_ATOMS;
    const int nn = act ? n : 0;
    const float4* Q4 = (const float4*)qkv;     // rows of 48 float4 (192 f32)
    const float4* V4 = (const float4*)vvals;   // 3 planes of PLANE/4 float4
    const float4* W4 = (const float4*)wdir_l;

    const float4 q0 = Q4[nn * 48 + h4];
    const float4 q1 = Q4[nn * 48 + h4 + 1];
    const float4 wd0 = W4[h4];
    const float4 wd1 = W4[h4 + 1];

    int   jl[KNN];
    float logits[KNN];
#pragma unroll
    for (int kk = 0; kk < KNN; ++kk) {
        const int e = nn * KNN + kk;
        const int j = nbr[e];
        jl[kk] = j;
        const float4 k0 = Q4[j * 48 + 16 + h4];
        const float4 k1 = Q4[j * 48 + 16 + h4 + 1];
        const float dot = q0.x * k0.x + q0.y * k0.y + q0.z * k0.z + q0.w * k0.w +
                          q1.x * k1.x + q1.y * k1.y + q1.z * k1.z + q1.w * k1.w;
        logits[kk] = dot * 0.35355339059f + dist_bias_l[bins[e] * HEADS + head];
    }
    float mx = logits[0];
#pragma unroll
    for (int kk = 1; kk < KNN; ++kk) mx = fmaxf(mx, logits[kk]);
    float ssum = 0.0f;
#pragma unroll
    for (int kk = 0; kk < KNN; ++kk) { logits[kk] = __expf(logits[kk] - mx); ssum += logits[kk]; }
    const float inv = 1.0f / ssum;

    float4 a0 = {0, 0, 0, 0}, a1 = {0, 0, 0, 0};
    float4 v00 = {0, 0, 0, 0}, v01 = {0, 0, 0, 0};
    float4 v10 = {0, 0, 0, 0}, v11 = {0, 0, 0, 0};
    float4 v20 = {0, 0, 0, 0}, v21 = {0, 0, 0, 0};

#pragma unroll
    for (int kk = 0; kk < KNN; ++kk) {
        const int e = nn * KNN + kk;
        const int j = jl[kk];
        const float p = logits[kk] * inv;
        const float dx = dirn[e * 3 + 0];
        const float dy = dirn[e * 3 + 1];
        const float dz = dirn[e * 3 + 2];
        const float4 vn0 = Q4[j * 48 + 32 + h4];
        const float4 vn1 = Q4[j * 48 + 32 + h4 + 1];
        mad4(a0, p, vn0);
        mad4(a1, p, vn1);
        const int jb = j * 16 + h4;
        const float4 x00 = V4[jb],                 x01 = V4[jb + 1];
        const float4 x10 = V4[PLANE / 4 + jb],     x11 = V4[PLANE / 4 + jb + 1];
        const float4 x20 = V4[PLANE / 2 + jb],     x21 = V4[PLANE / 2 + jb + 1];
        madd4(v00, p, x00, wd0, dx); madd4(v01, p, x01, wd1, dx);
        madd4(v10, p, x10, wd0, dy); madd4(v11, p, x11, wd1, dy);
        madd4(v20, p, x20, wd0, dz); madd4(v21, p, x21, wd1, dz);
    }

    const float av[8] = {a0.x, a0.y, a0.z, a0.w, a1.x, a1.y, a1.z, a1.w};
    const int mb = aswz(n, hd8, 2);
#pragma unroll
    for (int d = 0; d < 8; ++d)
        msg16[mb + d] = act ? (_Float16)av[d] : (_Float16)0.0f;

    if (act) {
        float4* VB = (float4*)vbuf;
        const int ob = n * 16 + h4;
        float4 t;
        t = VB[ob];                 mad4(t, 1.0f, v00); VB[ob] = t;
        t = VB[ob + 1];             mad4(t, 1.0f, v01); VB[ob + 1] = t;
        t = VB[PLANE / 4 + ob];     mad4(t, 1.0f, v10); VB[PLANE / 4 + ob] = t;
        t = VB[PLANE / 4 + ob + 1]; mad4(t, 1.0f, v11); VB[PLANE / 4 + ob + 1] = t;
        t = VB[PLANE / 2 + ob];     mad4(t, 1.0f, v20); VB[PLANE / 2 + ob] = t;
        t = VB[PLANE / 2 + ob + 1]; mad4(t, 1.0f, v21); VB[PLANE / 2 + ob + 1] = t;
    }
}

// v *= sigmoid-gate; emits A-swizzled f16 copy. Thread per (plane, atom, 8-ch group)
__global__ void gate_kernel(float* __restrict__ vbuf, const float* __restrict__ g,
                            _Float16* __restrict__ v16) {
    int idx = blockIdx.x * blockDim.x + threadIdx.x;
    if (idx >= 3 * NPAD * 8) return;
    const int c = idx / (NPAD * 8);
    const int r = idx - c * (NPAD * 8);
    const int n = r >> 3, gg = r & 7;
    float4* VB = (float4*)(vbuf + (size_t)c * PLANE + n * 64 + gg * 8);
    const float4* G4 = (const float4*)(g + n * 64 + gg * 8);
    float4 a = VB[0], b = VB[1];
    const float4 g0 = G4[0], g1 = G4[1];
    a.x *= g0.x; a.y *= g0.y; a.z *= g0.z; a.w *= g0.w;
    b.x *= g1.x; b.y *= g1.y; b.z *= g1.z; b.w *= g1.w;
    VB[0] = a; VB[1] = b;
    _Float16* dst = v16 + (size_t)c * PLANE + aswz(n, gg * 8, 2);
    dst[0] = (_Float16)a.x; dst[1] = (_Float16)a.y;
    dst[2] = (_Float16)a.z; dst[3] = (_Float16)a.w;
    dst[4] = (_Float16)b.x; dst[5] = (_Float16)b.y;
    dst[6] = (_Float16)b.z; dst[7] = (_Float16)b.w;
}

// ---------------------------------------------------------------------------
// residue mean-pool + linear head
// ---------------------------------------------------------------------------
__global__ __launch_bounds__(256) void head_kernel(const float* __restrict__ s,
                                                   const float* __restrict__ Wout,
                                                   const float* __restrict__ bout,
                                                   float* __restrict__ out) {
    const int r = blockIdx.x * 8 + (threadIdx.x >> 5);
    const int lane = threadIdx.x & 31;
    if (r >= RRES) return;
    float p0 = 0.f, p1 = 0.f;
    for (int t = 0; t < APR; ++t) {
        const int a = r * APR + t;
        p0 += s[a * 64 + lane];
        p1 += s[a * 64 + 32 + lane];
    }
    float o0 = p0 * Wout[lane * 2 + 0] + p1 * Wout[(lane + 32) * 2 + 0];
    float o1 = p0 * Wout[lane * 2 + 1] + p1 * Wout[(lane + 32) * 2 + 1];
#pragma unroll
    for (int mk = 16; mk >= 1; mk >>= 1) {
        o0 += __shfl_xor(o0, mk, 32);
        o1 += __shfl_xor(o1, mk, 32);
    }
    if (lane == 0) {
        out[r * 2 + 0] = o0 * (1.0f / APR) + bout[0];
        out[r * 2 + 1] = o1 * (1.0f / APR) + bout[1];
    }
}

// ---------------------------------------------------------------------------
extern "C" void kernel_launch(void* const* d_in, const int* in_sizes, int n_in,
                              void* d_out, int out_size, void* d_ws, size_t ws_size,
                              hipStream_t stream) {
    const float* coords    = (const float*)d_in[0];
    const int*   atom_ids  = (const int*)d_in[1];
    const int*   res_ids   = (const int*)d_in[2];
    const int*   elem_ids  = (const int*)d_in[3];
    const float* emb_atom  = (const float*)d_in[5];
    const float* emb_res   = (const float*)d_in[6];
    const float* emb_elem  = (const float*)d_in[7];
    const float* Win       = (const float*)d_in[8];
    const float* Wq        = (const float*)d_in[9];
    const float* Wk        = (const float*)d_in[10];
    const float* Wv        = (const float*)d_in[11];
    const float* Wvec      = (const float*)d_in[12];
    const float* wdir      = (const float*)d_in[13];
    const float* dist_bias = (const float*)d_in[14];
    const float* Wo        = (const float*)d_in[15];
    const float* W1        = (const float*)d_in[16];
    const float* W2        = (const float*)d_in[17];
    const float* Wg        = (const float*)d_in[18];
    const float* Wout      = (const float*)d_in[19];
    const float* bout      = (const float*)d_in[20];
    float* out = (float*)d_out;

    // ---- workspace carve ----
    char* ws = (char*)d_ws;
    size_t off = 0;
    auto alloc = [&](size_t bytes) -> void* {
        void* p = ws + off;
        off = (off + bytes + 255) & ~(size_t)255;
        return p;
    };
    _Float16* Win16  = (_Float16*)alloc(4096 * 2);
    _Float16* Wqkv16 = (_Float16*)alloc((size_t)LAYERS * 12288 * 2);
    _Float16* Wvec16 = (_Float16*)alloc((size_t)LAYERS * 4096 * 2);
    _Float16* Wo16   = (_Float16*)alloc((size_t)LAYERS * 4096 * 2);
    _Float16* W116   = (_Float16*)alloc((size_t)LAYERS * 8192 * 2);
    _Float16* W216   = (_Float16*)alloc((size_t)LAYERS * 8192 * 2);
    _Float16* Wg16   = (_Float16*)alloc((size_t)LAYERS * 4096 * 2);
    float*    qkvcat = (float*)alloc((size_t)LAYERS * 12288 * 4);
    _Float16* feat16 = (_Float16*)alloc((size_t)PLANE * 2);
    _Float16* h16    = (_Float16*)alloc((size_t)PLANE * 2);
    _Float16* msg16  = (_Float16*)alloc((size_t)PLANE * 2);
    _Float16* ffn16  = (_Float16*)alloc((size_t)NPAD * 128 * 2);
    _Float16* v16    = (_Float16*)alloc((size_t)3 * PLANE * 2);
    float* sbuf  = (float*)alloc((size_t)PLANE * 4);
    float* qkv   = (float*)alloc((size_t)NPAD * 192 * 4);
    float* vvals = (float*)alloc((size_t)3 * PLANE * 4);
    float* vbuf  = (float*)alloc((size_t)3 * PLANE * 4);
    float* gbuf  = (float*)alloc((size_t)PLANE * 4);
    int*   nbr   = (int*)alloc((size_t)N_ATOMS * KNN * 4);
    int*   bins  = (int*)alloc((size_t)N_ATOMS * KNN * 4);
    float* dirn  = (float*)alloc((size_t)N_ATOMS * KNN * 3 * 4);
    float* pd2   = (float*)alloc((size_t)N_ATOMS * NCHUNK * KNN * 4);
    int*   pidx  = (int*)alloc((size_t)N_ATOMS * NCHUNK * KNN * 4);

    // ---- weight prep: f32 -> fragment-ordered f16 ----
    qkvcat_kernel<<<(LAYERS * 12288 + 255) / 256, 256, 0, stream>>>(Wq, Wk, Wv, qkvcat);
    wswz_kernel<<<(4096 + 255) / 256, 256, 0, stream>>>(Win, Win16, 64, 64, 1);
    wswz_kernel<<<(LAYERS * 12288 + 255) / 256, 256, 0, stream>>>(qkvcat, Wqkv16, 64, 192, LAYERS);
    wswz_kernel<<<(LAYERS * 4096 + 255) / 256, 256, 0, stream>>>(Wvec, Wvec16, 64, 64, LAYERS);
    wswz_kernel<<<(LAYERS * 4096 + 255) / 256, 256, 0, stream>>>(Wo, Wo16, 64, 64, LAYERS);
    wswz_kernel<<<(LAYERS * 8192 + 255) / 256, 256, 0, stream>>>(W1, W116, 64, 128, LAYERS);
    wswz_kernel<<<(LAYERS * 8192 + 255) / 256, 256, 0, stream>>>(W2, W216, 128, 64, LAYERS);
    wswz_kernel<<<(LAYERS * 4096 + 255) / 256, 256, 0, stream>>>(Wg, Wg16, 64, 64, LAYERS);

    // ---- graph build ----
    embed_kernel<<<(NPAD * 8) / 256, 256, 0, stream>>>(atom_ids, res_ids, elem_ids,
                                                       emb_atom, emb_res, emb_elem, feat16);
    {
        dim3 g(NPAD / 256, NCHUNK);
        knn_partial_kernel<<<g, 256, 0, stream>>>(coords, pd2, pidx);
    }
    knn_merge_kernel<<<(N_ATOMS + 255) / 256, 256, 0, stream>>>(pd2, pidx, nbr);
    edges_kernel<<<(N_ATOMS * KNN + 255) / 256, 256, 0, stream>>>(coords, nbr, dirn, bins);

    // ---- lift: s = feat @ Win ; v = 0 ; vvals(l=0) = 0 ----
    gemm_wmma_kernel<64, 64, EPI_NONE><<<NPAD / 64, 128, 0, stream>>>(feat16, Win16, sbuf);
    hipMemsetAsync(vbuf, 0, (size_t)3 * PLANE * 4, stream);
    hipMemsetAsync(vvals, 0, (size_t)3 * PLANE * 4, stream);

    for (int l = 0; l < LAYERS; ++l) {
        ln16_kernel<<<NPAD / 256, 256, 0, stream>>>(sbuf, h16);
        gemm_wmma_kernel<64, 192, EPI_NONE><<<NPAD / 64, 384, 0, stream>>>(
            h16, Wqkv16 + (size_t)l * 12288, qkv);
        if (l > 0) {
            for (int c = 0; c < 3; ++c)
                gemm_wmma_kernel<64, 64, EPI_NONE><<<NPAD / 64, 128, 0, stream>>>(
                    v16 + (size_t)c * PLANE, Wvec16 + (size_t)l * 4096,
                    vvals + (size_t)c * PLANE);
        }
        attn_kernel<<<NPAD / 32, 256, 0, stream>>>(qkv, nbr, bins, dirn,
                                                   dist_bias + (size_t)l * BINS * HEADS,
                                                   wdir + (size_t)l * HDIM,
                                                   vvals, vbuf, msg16);
        gemm_wmma_kernel<64, 64, EPI_ACC><<<NPAD / 64, 128, 0, stream>>>(
            msg16, Wo16 + (size_t)l * 4096, sbuf);
        ln16_kernel<<<NPAD / 256, 256, 0, stream>>>(sbuf, h16);
        gemm_wmma_kernel<64, 128, EPI_GELU16><<<NPAD / 64, 256, 0, stream>>>(
            h16, W116 + (size_t)l * 8192, ffn16);
        gemm_wmma_kernel<128, 64, EPI_ACC><<<NPAD / 64, 128, 0, stream>>>(
            ffn16, W216 + (size_t)l * 8192, sbuf);
        gemm_wmma_kernel<64, 64, EPI_SIG><<<NPAD / 64, 128, 0, stream>>>(
            h16, Wg16 + (size_t)l * 4096, gbuf);
        gate_kernel<<<(3 * NPAD * 8 + 255) / 256, 256, 0, stream>>>(vbuf, gbuf, v16);
    }

    head_kernel<<<(RRES + 7) / 8, 256, 0, stream>>>(sbuf, Wout, bout, out);
}
